// Uni2TwoUpPropPred_2259152797782
// MI455X (gfx1250) — compile-verified, matrix operand order checked by reading, not run on hardware
//
#include <hip/hip_runtime.h>
#include <hip/hip_bf16.h>

#define NN   8192
#define NE   131072
#define NB   64
#define HID  128
#define HEADS 16
#define DH   8
#define NRBF 50

typedef _Float16 v8h  __attribute__((ext_vector_type(8)));
typedef _Float16 v16h __attribute__((ext_vector_type(16)));
typedef float    v8f  __attribute__((ext_vector_type(8)));

__device__ __forceinline__ v16h cat16h(v8h lo, v8h hi) {
  return __builtin_shufflevector(lo, hi, 0,1,2,3,4,5,6,7,8,9,10,11,12,13,14,15);
}

#define WMMA_F16(a, b, c) \
  __builtin_amdgcn_wmma_f32_16x16x32_f16(false, (a), false, (b), (short)0, (c), false, false)

// CDNA5 async VMEM->LDS copy (ASYNCcnt-tracked), 16 bytes per lane.
// ldsOff: LDS byte address (low 32 bits of the shared-memory flat address).
__device__ __forceinline__ void async_load_b128(unsigned ldsOff, const void* gptr)
{
  unsigned long long ga = (unsigned long long)gptr;
  asm volatile("global_load_async_to_lds_b128 %0, %1, off"
               :: "v"(ldsOff), "v"(ga) : "memory");
}
__device__ __forceinline__ void wait_async0()
{
  asm volatile("s_wait_asynccnt 0x0" ::: "memory");
}

// ---------------------------------------------------------------------------
// Generic 2-layer MLP:  Y = relu(X @ W1 + b1) @ W2 + b2 (+resid)
// X: [M x K1] f16 row-major (ldx = K1, K1 % 32 == 0, zero padded)
// W1t: [128 x K1] f16 (transposed, row = output col), W2t: [128 x 128] f16
// Block: 256 threads = 8 waves, computes 128 rows x 128 cols.
// Wave w owns rows w*16..w*16+16, all 128 cols (8 WMMA accumulators).
// Weight chunks staged to LDS via global_load_async_to_lds_b128.
// ---------------------------------------------------------------------------
__global__ __launch_bounds__(256)
void mlp2_kernel(const _Float16* __restrict__ X, int ldx, int K1,
                 const _Float16* __restrict__ W1t, const float* __restrict__ b1,
                 const _Float16* __restrict__ W2t, const float* __restrict__ b2,
                 float* __restrict__ Yf32, _Float16* __restrict__ Yf16,
                 const float* __restrict__ resid)
{
  __shared__ _Float16 Wls[128 * 32];   // weight chunk [n][k]   (8 KB)
  __shared__ _Float16 Hs[128 * 128];   // hidden tile, relu'd   (32 KB)

  const int tid   = threadIdx.x;
  const int lane  = tid & 31;
  const int wave  = tid >> 5;          // row tile 0..7
  const int r     = lane & 15;
  const int halfk = (lane >> 4) * 8;   // K-offset of this half-wave
  const int mD    = (lane >> 4) * 8;   // D-fragment row offset
  const long m0   = (long)blockIdx.x * 128 + wave * 16;

  // staging assignment: thread copies 8 f16 (16 bytes) of the [128 x 32] chunk
  const int sn = tid >> 1;             // output column 0..127
  const int sk = (tid & 1) * 16;       // k sub-chunk 0 or 16
  const unsigned wlsOff = (unsigned)(size_t)(Wls + sn * 32 + sk);

  v8f acc[8] = {v8f{}, v8f{}, v8f{}, v8f{}, v8f{}, v8f{}, v8f{}, v8f{}};

  for (int kc = 0; kc < K1; kc += 32) {
    __syncthreads();                       // prior Wls reads complete
    async_load_b128(wlsOff,      W1t + (size_t)sn * K1 + kc + sk);
    // A fragment straight from global (16-bit A layout)
    const _Float16* ap = X + (size_t)(m0 + r) * ldx + kc + halfk;
    v16h a = cat16h(*(const v8h*)ap, *(const v8h*)(ap + 16));
    if (kc + 32 < K1) __builtin_prefetch(ap + 32, 0, 0);
    wait_async0();
    __syncthreads();                       // chunk visible to all waves
#pragma unroll
    for (int j = 0; j < 8; ++j) {
      const _Float16* bp = Wls + (j * 16 + r) * 32 + halfk;
      v16h bf = cat16h(*(const v8h*)bp, *(const v8h*)(bp + 16));
      acc[j] = WMMA_F16(a, bf, acc[j]);
    }
  }

  __syncthreads();
  // bias + relu -> Hs (f16)
#pragma unroll
  for (int j = 0; j < 8; ++j) {
    int n = j * 16 + r;
    float bb = b1[n];
#pragma unroll
    for (int v = 0; v < 8; ++v) {
      float x = acc[j][v] + bb;
      x = x > 0.0f ? x : 0.0f;
      Hs[(wave * 16 + mD + v) * 128 + n] = (_Float16)x;
    }
  }
  __syncthreads();

  v8f acc2[8] = {v8f{}, v8f{}, v8f{}, v8f{}, v8f{}, v8f{}, v8f{}, v8f{}};
  for (int kc = 0; kc < 128; kc += 32) {
    __syncthreads();
    async_load_b128(wlsOff, W2t + (size_t)sn * 128 + kc + sk);
    const _Float16* ap = Hs + (wave * 16 + r) * 128 + kc + halfk;
    v16h a = cat16h(*(const v8h*)ap, *(const v8h*)(ap + 16));
    wait_async0();
    __syncthreads();
#pragma unroll
    for (int j = 0; j < 8; ++j) {
      const _Float16* bp = Wls + (j * 16 + r) * 32 + halfk;
      v16h bf = cat16h(*(const v8h*)bp, *(const v8h*)(bp + 16));
      acc2[j] = WMMA_F16(a, bf, acc2[j]);
    }
  }

  // epilogue: bias (+resid), store f32 and/or f16
#pragma unroll
  for (int j = 0; j < 8; ++j) {
    int n = j * 16 + r;
    float bb = b2[n];
#pragma unroll
    for (int v = 0; v < 8; ++v) {
      long m = m0 + mD + v;
      float y = acc2[j][v] + bb;
      if (resid) y += resid[(size_t)m * 128 + n];
      if (Yf32)  Yf32[(size_t)m * 128 + n] = y;
      if (Yf16)  Yf16[(size_t)m * 128 + n] = (_Float16)y;
    }
  }
}

// ---------------------------------------------------------------------------
// Weight transpose + f32->f16 convert: Wt[n][k] = (k < K) ? W[k*Nout + n] : 0
// ---------------------------------------------------------------------------
__global__ void transposeW_kernel(const float* __restrict__ W, _Float16* __restrict__ Wt,
                                  int K, int Nout, int Kpad)
{
  int i = blockIdx.x * 256 + threadIdx.x;
  if (i >= Nout * Kpad) return;
  int n = i / Kpad, k = i - n * Kpad;
  Wt[i] = (k < K) ? (_Float16)W[(size_t)k * Nout + n] : (_Float16)0.0f;
}

__global__ void fill_f32_kernel(float* p, float v, int n)
{
  int i = blockIdx.x * 256 + threadIdx.x;
  if (i < n) p[i] = v;
}

__global__ void copy_f32_kernel(const float* __restrict__ s, float* __restrict__ d, int n)
{
  int i = blockIdx.x * 256 + threadIdx.x;
  if (i < n) d[i] = s[i];
}

// node featurization + embedding: h = feat @ Wn + bn  (K=15, trivial)
__global__ void node_embed_kernel(const int* __restrict__ node_type,
                                  const float* __restrict__ remap,
                                  const float* __restrict__ Wn, const float* __restrict__ bn,
                                  float* __restrict__ h, _Float16* __restrict__ h16)
{
  int i = blockIdx.x * 256 + threadIdx.x;
  if (i >= NN * HID) return;
  int n = i >> 7, c = i & 127;
  float q = (float)node_type[n] * (1.0f / 9.0f);
  float p0 = 1.0f, p1 = q, p2 = q * q;
  float s = bn[c];
#pragma unroll
  for (int t = 0; t < 5; ++t) {
    float r = remap[n * 5 + t];
    s += r * (p0 * Wn[(t * 3 + 0) * HID + c] +
              p1 * Wn[(t * 3 + 1) * HID + c] +
              p2 * Wn[(t * 3 + 2) * HID + c]);
  }
  h[i] = s;
  h16[i] = (_Float16)s;
}

// per-edge layer-invariant KV columns: [0:128) edge_emb, [128:178) rbf, pad [434:448)
__global__ __launch_bounds__(256)
void edge_base_kernel(const float* __restrict__ pos, const int* __restrict__ ei,
                      const int* __restrict__ etype, const float* __restrict__ edge_emb,
                      _Float16* __restrict__ kv16)
{
  int e = blockIdx.x;
  int c = threadIdx.x;
  _Float16* row = kv16 + (size_t)e * 448;
  if (c < 128) {
    row[c] = (_Float16)edge_emb[(size_t)etype[e] * HID + c];
  } else if (c < 128 + NRBF) {
    int s = ei[e], d = ei[NE + e];
    float dx = pos[d * 3 + 0] - pos[s * 3 + 0];
    float dy = pos[d * 3 + 1] - pos[s * 3 + 1];
    float dz = pos[d * 3 + 2] - pos[s * 3 + 2];
    float dist = sqrtf(dx * dx + dy * dy + dz * dz);
    const float dr = 10.0f / 49.0f;
    float t = dist - (float)(c - 128) * dr;
    row[c] = (_Float16)__expf((-0.5f / (dr * dr)) * t * t);
  } else if (c >= 192 && c < 192 + 14) {
    row[434 + (c - 192)] = (_Float16)0.0f;   // zero-pad 434..448
  }
}

// per-layer: kv[:,178:306) = h[dst], kv[:,306:434) = h[src]
__global__ __launch_bounds__(128)
void scatter_h_kernel(const int* __restrict__ ei, const _Float16* __restrict__ h16,
                      _Float16* __restrict__ kv16)
{
  int e = blockIdx.x, c = threadIdx.x;
  int s = ei[e], d = ei[NE + e];
  _Float16* row = kv16 + (size_t)e * 448;
  row[178 + c] = h16[(size_t)d * HID + c];
  row[306 + c] = h16[(size_t)s * HID + c];
}

__device__ __forceinline__ void atomicMaxF(float* a, float v)
{
  int* ai = (int*)a;
  int old = __float_as_int(*a);
  while (v > __int_as_float(old)) {
    int prev = atomicCAS(ai, old, __float_as_int(v));
    if (prev == old) break;
    old = prev;
  }
}

// logits = (q[dst] . k) * scale per head; running per-node max
__global__ void logits_kernel(const _Float16* __restrict__ q16, const _Float16* __restrict__ k16,
                              const int* __restrict__ ei, float* __restrict__ logits,
                              float* __restrict__ nodemax)
{
  int i = blockIdx.x * 256 + threadIdx.x;
  if (i >= NE * HEADS) return;
  int e = i >> 4, hh = i & 15;
  int d = ei[NE + e];
  const _Float16* qp = q16 + (size_t)d * HID + hh * DH;
  const _Float16* kp = k16 + (size_t)e * HID + hh * DH;
  float s = 0.0f;
#pragma unroll
  for (int j = 0; j < DH; ++j) s += (float)qp[j] * (float)kp[j];
  s *= 0.35355339059f;  // 1/sqrt(8)
  logits[i] = s;
  atomicMaxF(&nodemax[(size_t)d * HEADS + hh], s);
}

__global__ void expsum_kernel(const float* __restrict__ logits, const int* __restrict__ ei,
                              const float* __restrict__ nodemax, float* __restrict__ alpha,
                              float* __restrict__ denom)
{
  int i = blockIdx.x * 256 + threadIdx.x;
  if (i >= NE * HEADS) return;
  int e = i >> 4, hh = i & 15;
  int d = ei[NE + e];
  float a = __expf(logits[i] - nodemax[(size_t)d * HEADS + hh]);
  alpha[i] = a;
  atomicAdd(&denom[(size_t)d * HEADS + hh], a);
}

__global__ void agg_kernel(const int* __restrict__ ei, const float* __restrict__ alpha,
                           const float* __restrict__ denom, const _Float16* __restrict__ v16a,
                           float* __restrict__ agg)
{
  long i = (long)blockIdx.x * 256 + threadIdx.x;
  if (i >= (long)NE * HID) return;
  int e = (int)(i >> 7), c = (int)(i & 127), hh = c >> 3;
  int d = ei[NE + e];
  float w = alpha[(size_t)e * HEADS + hh] / denom[(size_t)d * HEADS + hh];
  atomicAdd(&agg[(size_t)d * HID + c], w * (float)v16a[(size_t)e * HID + c]);
}

__global__ void cat_kernel(const float* __restrict__ agg, const _Float16* __restrict__ h16,
                           _Float16* __restrict__ cat16)
{
  int i = blockIdx.x * 256 + threadIdx.x;
  if (i >= NN * 256) return;
  int n = i >> 8, c = i & 255;
  cat16[i] = (c < 128) ? (_Float16)agg[(size_t)n * HID + c]
                       : h16[(size_t)n * HID + (c - 128)];
}

// pre_out MLP (128->128 relu ->1) + batch sum readout
__global__ __launch_bounds__(128)
void preout_kernel(const float* __restrict__ h, const float* __restrict__ W1,
                   const float* __restrict__ b1, const float* __restrict__ W2,
                   const float* __restrict__ b2, const int* __restrict__ batch,
                   float* __restrict__ outB)
{
  __shared__ float hs[128];
  __shared__ float red[128];
  int n = blockIdx.x, c = threadIdx.x;
  hs[c] = h[(size_t)n * HID + c];
  __syncthreads();
  float s = b1[c];
  for (int k = 0; k < 128; ++k) s += hs[k] * W1[(size_t)k * HID + c];
  s = fmaxf(s, 0.0f) * W2[c];
  red[c] = s;
  __syncthreads();
  for (int st = 64; st > 0; st >>= 1) {
    if (c < st) red[c] += red[c + st];
    __syncthreads();
  }
  if (c == 0) atomicAdd(&outB[batch[n]], red[0] + b2[0]);
}

// ---------------------------------------------------------------------------
extern "C" void kernel_launch(void* const* d_in, const int* in_sizes, int n_in,
                              void* d_out, int out_size, void* d_ws, size_t ws_size,
                              hipStream_t stream)
{
  (void)in_sizes; (void)n_in; (void)out_size; (void)ws_size;

  const int*   node_type = (const int*)d_in[0];
  const float* remap     = (const float*)d_in[1];
  const float* pos       = (const float*)d_in[2];
  const int*   ei        = (const int*)d_in[3];
  const int*   etype     = (const int*)d_in[4];
  const int*   batch     = (const int*)d_in[5];
  // params flattened (jax sorted-dict order): edge_emb, layers[4]{hk,hq,hv,node_out}x(W1,b1,W2,b2), node_emb(W,b), pre_out(W1,b1,W2,b2)
  const float* edge_emb  = (const float*)d_in[6];
  auto F = [&](int i) { return (const float*)d_in[i]; };
  const float* Wn = F(71);  const float* bn = F(72);
  const float* poW1 = F(73); const float* pob1 = F(74);
  const float* poW2 = F(75); const float* pob2 = F(76);

  char* ws = (char*)d_ws;
  size_t off = 0;
  auto alloc = [&](size_t bytes) -> char* {
    char* p = ws + off;
    off = (off + bytes + 255) & ~(size_t)255;
    return p;
  };
  float*    h     = (float*)alloc((size_t)NN * HID * 4);
  _Float16* h16   = (_Float16*)alloc((size_t)NN * HID * 2);
  _Float16* q16   = (_Float16*)alloc((size_t)NN * HID * 2);
  _Float16* k16   = (_Float16*)alloc((size_t)NE * HID * 2);
  _Float16* v16a  = (_Float16*)alloc((size_t)NE * HID * 2);
  _Float16* kv16  = (_Float16*)alloc((size_t)NE * 448 * 2);
  float*    logits= (float*)alloc((size_t)NE * HEADS * 4);
  float*    alpha = (float*)alloc((size_t)NE * HEADS * 4);
  float*    nodemax = (float*)alloc((size_t)NN * HEADS * 4);
  float*    denom = (float*)alloc((size_t)NN * HEADS * 4);
  float*    agg   = (float*)alloc((size_t)NN * HID * 4);
  _Float16* cat16 = (_Float16*)alloc((size_t)NN * 256 * 2);

  // transposed f16 weights: per layer {hkW1t, hkW2t, hqW1t, hqW2t, hvW1t, hvW2t, noW1t, noW2t}
  _Float16* Wt[4][8];
  const int k1pad[8] = {448, 128, 128, 128, 448, 128, 256, 128};
  for (int L = 0; L < 4; ++L)
    for (int m = 0; m < 8; ++m)
      Wt[L][m] = (_Float16*)alloc((size_t)128 * k1pad[m] * 2);

  float* outB = (float*)d_out;          // [64]
  float* outPos = (float*)d_out + NB;   // [8192*3]

  // --- output init + pos passthrough ---
  fill_f32_kernel<<<1, 256, 0, stream>>>(outB, 0.0f, NB);
  copy_f32_kernel<<<(NN * 3 + 255) / 256, 256, 0, stream>>>(pos, outPos, NN * 3);

  // --- weight prep (f16 transpose, K zero-padded) ---
  for (int L = 0; L < 4; ++L) {
    int base = 7 + L * 16;
    const float* W1s[4] = {F(base + 0), F(base + 4), F(base + 8), F(base + 12)};
    const float* W2s[4] = {F(base + 2), F(base + 6), F(base + 10), F(base + 14)};
    const int    kreal[4] = {434, 128, 434, 256};
    const int    kpads[4] = {448, 128, 448, 256};
    for (int m = 0; m < 4; ++m) {
      int nelem = 128 * kpads[m];
      transposeW_kernel<<<(nelem + 255) / 256, 256, 0, stream>>>(
          W1s[m], Wt[L][m * 2 + 0], kreal[m], 128, kpads[m]);
      transposeW_kernel<<<(128 * 128 + 255) / 256, 256, 0, stream>>>(
          W2s[m], Wt[L][m * 2 + 1], 128, 128, 128);
    }
  }

  // --- featurization ---
  node_embed_kernel<<<(NN * HID + 255) / 256, 256, 0, stream>>>(node_type, remap, Wn, bn, h, h16);
  edge_base_kernel<<<NE, 256, 0, stream>>>(pos, ei, etype, edge_emb, kv16);

  // --- layers ---
  for (int L = 0; L < 4; ++L) {
    int base = 7 + L * 16;
    const float* hk_b1 = F(base + 1);  const float* hk_b2 = F(base + 3);
    const float* hq_b1 = F(base + 5);  const float* hq_b2 = F(base + 7);
    const float* hv_b1 = F(base + 9);  const float* hv_b2 = F(base + 11);
    const float* no_b1 = F(base + 13); const float* no_b2 = F(base + 15);

    scatter_h_kernel<<<NE, 128, 0, stream>>>(ei, h16, kv16);

    // q = MLP(h)
    mlp2_kernel<<<NN / 128, 256, 0, stream>>>(h16, 128, 128,
        Wt[L][2], hq_b1, Wt[L][3], hq_b2, nullptr, q16, nullptr);
    // k = MLP(kv)
    mlp2_kernel<<<NE / 128, 256, 0, stream>>>(kv16, 448, 448,
        Wt[L][0], hk_b1, Wt[L][1], hk_b2, nullptr, k16, nullptr);
    // v = MLP(kv)
    mlp2_kernel<<<NE / 128, 256, 0, stream>>>(kv16, 448, 448,
        Wt[L][4], hv_b1, Wt[L][5], hv_b2, nullptr, v16a, nullptr);

    // attention: scatter softmax over dst
    fill_f32_kernel<<<(NN * HEADS + 255) / 256, 256, 0, stream>>>(nodemax, -1e30f, NN * HEADS);
    fill_f32_kernel<<<(NN * HEADS + 255) / 256, 256, 0, stream>>>(denom, 0.0f, NN * HEADS);
    fill_f32_kernel<<<(NN * HID + 255) / 256, 256, 0, stream>>>(agg, 0.0f, NN * HID);
    logits_kernel<<<(NE * HEADS + 255) / 256, 256, 0, stream>>>(q16, k16, ei, logits, nodemax);
    expsum_kernel<<<(NE * HEADS + 255) / 256, 256, 0, stream>>>(logits, ei, nodemax, alpha, denom);
    agg_kernel<<<(int)(((long)NE * HID + 255) / 256), 256, 0, stream>>>(ei, alpha, denom, v16a, agg);

    // node_out: h = MLP([agg, h]) + h  (in-place safe: resid read/write same thread)
    cat_kernel<<<(NN * 256 + 255) / 256, 256, 0, stream>>>(agg, h16, cat16);
    mlp2_kernel<<<NN / 128, 256, 0, stream>>>(cat16, 256, 256,
        Wt[L][6], no_b1, Wt[L][7], no_b2, h, h16, h);
  }

  // --- readout ---
  preout_kernel<<<NN, 128, 0, stream>>>(h, poW1, pob1, poW2, pob2, batch, outB);
}
